// RNN_79740362817939
// MI455X (gfx1250) — matrix operand order repaired and checked by reading, not compiled, not run on hardware
//
#include <hip/hip_runtime.h>
#include <hip/hip_bf16.h>
#include <math.h>

// ---------------------------------------------------------------------------
// RNN forward_sequence for MI455X (gfx1250, wave32, WMMA).
// T=2048 B=64 I=512 H=1024 O=512, K = I+H = 1536.
// Persistent-grid design: 16 workgroups x 256 threads (8 waves), device-scope
// atomic barrier between the 2047 sequential recurrent steps. Weights are
// converted once to bf16 in a WMMA-B-swizzled layout (L2-resident, 4.5 MB).
// comb = [x_t, h_t] staged as bf16 in LDS each step (padded rows, 193 KB);
// the h part uses CDNA5 GLOBAL_LOAD_ASYNC_TO_LDS_B128 (ASYNCcnt).
// Math: v_wmma_f32_16x16x32_bf16, f32 accumulate.
// ---------------------------------------------------------------------------

typedef __attribute__((ext_vector_type(16))) __bf16 v16bf;
typedef __attribute__((ext_vector_type(8)))  __bf16 v8bf;
typedef __attribute__((ext_vector_type(8)))  float  v8f;

#define T_STEPS 2048
#define BATCH   64
#define IN_DIM  512
#define HID     1024
#define OUT_DIM 512
#define KTOT    1536          // IN_DIM + HID
#define KBLKS   48            // KTOT / 32
#define NWG     16
#define KP      1544          // padded LDS row stride (bf16 elements): no bank conflicts

#define SHUF16(a, b) __builtin_shufflevector((a), (b), 0,1,2,3,4,5,6,7,8,9,10,11,12,13,14,15)

// workspace layout (bytes)
#define WS_WH_OFF   0u                                   // bf16 [48][64][32][16]  = 3 MB
#define WS_WO_OFF   (3145728u)                           // bf16 [48][32][32][16]  = 1.5 MB
#define WS_H0_OFF   (3145728u + 1572864u)                // bf16 [64][1024]        = 128 KB
#define WS_H1_OFF   (WS_H0_OFF + 131072u)                // bf16 [64][1024]        = 128 KB
#define WS_SYNC_OFF (WS_H1_OFF + 131072u)                // u32 barrier counter

// ---------------------------------------------------------------------------
// Prep kernel: fp32 weights -> bf16 WMMA-B-swizzled blocks; zero h0; reset sync.
// Swizzled element idx = ((kblk*NBLK + nblk)*32 + lane)*16 + e
//   maps to W[kblk*32 + (lane<16 ? e : 16+e)][nblk*16 + (lane&15)]
// i.e. contiguous k = 16*half + e per lane, the same k order the A-fragment loads use.
// ---------------------------------------------------------------------------
__global__ void rnn_prep(const float* __restrict__ Wh, const float* __restrict__ Wo,
                         __bf16* __restrict__ wh_sw, __bf16* __restrict__ wo_sw,
                         __bf16* __restrict__ h0, unsigned* __restrict__ sync) {
  const long NWH = (long)KTOT * HID;       // 1,572,864
  const long NWO = (long)KTOT * OUT_DIM;   //   786,432
  const long NH0 = (long)BATCH * HID;      //    65,536
  long idx = (long)blockIdx.x * 256 + threadIdx.x;
  if (idx < NWH) {
    int e = (int)(idx & 15), l = (int)((idx >> 4) & 31);
    long blk = idx >> 9;
    int nblk = (int)(blk & 63), kblk = (int)(blk >> 6);
    int k = kblk * 32 + ((l < 16) ? e : 16 + e);
    int n = nblk * 16 + (l & 15);
    wh_sw[idx] = (__bf16)Wh[(long)k * HID + n];
  } else if (idx < NWH + NWO) {
    long j = idx - NWH;
    int e = (int)(j & 15), l = (int)((j >> 4) & 31);
    long blk = j >> 9;
    int nblk = (int)(blk & 31), kblk = (int)(blk >> 5);
    int k = kblk * 32 + ((l < 16) ? e : 16 + e);
    int n = nblk * 16 + (l & 15);
    wo_sw[j] = (__bf16)Wo[(long)k * OUT_DIM + n];
  } else if (idx < NWH + NWO + NH0) {
    h0[idx - NWH - NWO] = (__bf16)0.0f;
  } else if (idx == NWH + NWO + NH0) {
    *sync = 0u;
  }
}

// ---------------------------------------------------------------------------
// Stage comb = [bf16(x_t) | h_t] into LDS (row stride KP, bank-conflict padded).
// h part: CDNA5 async global->LDS copy (no VGPR round-trip, ASYNCcnt-tracked).
// NOTE: kernel has no static __shared__, so the dynamic-LDS block starts at
// LDS byte offset 0; async-LDS destination operands are raw LDS offsets.
// ---------------------------------------------------------------------------
__device__ __forceinline__ void stage_comb(__bf16* __restrict__ comb,
                                           const float* __restrict__ xt,
                                           const __bf16* __restrict__ hsrc,
                                           int tid) {
  // h_t: 64x1024 bf16 -> comb[:, 512:1536], 16 B per async op, 32 ops/thread
  #pragma unroll 4
  for (int v = 0; v < 32; ++v) {
    int vidx = tid + (v << 8);                   // 0..8191
    int flat = vidx << 3;                        // bf16 element index
    int row = flat >> 10, col = flat & 1023;
    unsigned lds_off = (unsigned)(((unsigned)row * KP + 512u + (unsigned)col) * 2u);
    unsigned goff    = (unsigned)flat * 2u;      // byte offset from hsrc
    asm volatile("global_load_async_to_lds_b128 %0, %1, %2"
                 :: "v"(lds_off), "v"(goff), "s"(hsrc)
                 : "memory");
  }
  // x_t: 64x512 fp32 -> bf16 (conversion needs VALU) -> comb[:, 0:512]
  const float4* xs = (const float4*)xt;
  #pragma unroll 4
  for (int v = 0; v < 32; ++v) {
    int vidx = tid + (v << 8);                   // 0..8191
    int flat = vidx << 2;
    int row = flat >> 9, col = flat & 511;
    float4 xv = xs[vidx];
    __bf16* d = comb + (size_t)row * KP + col;
    d[0] = (__bf16)xv.x; d[1] = (__bf16)xv.y; d[2] = (__bf16)xv.z; d[3] = (__bf16)xv.w;
  }
  // all of this wave's async LDS writes must land before the s_barrier
  asm volatile("s_wait_asynccnt 0" ::: "memory");
}

// ---------------------------------------------------------------------------
// Persistent recurrent kernel.
// ---------------------------------------------------------------------------
__global__ void __launch_bounds__(256) rnn_persist(
    const float* __restrict__ x, const float* __restrict__ b_h,
    const float* __restrict__ b_o, const __bf16* __restrict__ wh_sw,
    const __bf16* __restrict__ wo_sw, __bf16* __restrict__ hbuf0,
    __bf16* __restrict__ hbuf1, unsigned* __restrict__ sync,
    float* __restrict__ out) {
  extern __shared__ __bf16 comb[];               // [64][KP] bf16, LDS offset 0
  const int wg   = blockIdx.x;                   // 0..15, owns 64 cols of H
  const int tid  = threadIdx.x;
  const int wave = tid >> 5;
  const int lane = tid & 31;
  const int half = lane >> 4;                    // 0: rows 0..7, 1: rows 8..15 of tile
  const int lm   = lane & 15;                    // output column within tile
  const int c0   = half ? 16 : 0;                // A-fragment K-chunk offsets (contiguous k = 16*half + e)
  const int c1   = half ? 24 : 8;

  // H-phase mapping: 8 waves -> 4 N-tiles x 2 (each wave does 2 stacked M-tiles)
  const int nt     = wave & 3;
  const int mts    = (wave >> 2) << 1;           // first of two M-tiles (0 or 2)
  const int ncol   = wg * 64 + nt * 16 + lm;     // global H column
  const int nblk_h = wg * 4 + nt;
  const float bvh  = b_h[ncol];

  // Per-lane constant base pointers: all k-block steps become immediate offsets
  // (stride 64*512 bf16 = 64 KiB/blk, 48 blks ~ 3 MB, inside signed 24-bit imm).
  const __bf16* wbase = wh_sw + ((size_t)nblk_h * 32 + lane) * 16;
  const __bf16* a0base = comb;                   // + (row)*KP, filled per use

  for (int t = 0; t < T_STEPS - 1; ++t) {
    const __bf16* hsrc = (t & 1) ? hbuf1 : hbuf0;
    stage_comb(comb, x + (size_t)t * BATCH * IN_DIM, hsrc, tid);
    // prefetch next timestep's x slab (256 threads * 512B = full 128 KB slab)
    const char* nx = (const char*)(x + (size_t)(t + 1) * BATCH * IN_DIM);
    __builtin_prefetch(nx + (size_t)tid * 512, 0, 0);
    __syncthreads();

    v8f acc0, acc1;
    #pragma unroll
    for (int r = 0; r < 8; ++r) { acc0[r] = bvh; acc1[r] = bvh; }

    const __bf16* ar0 = a0base + (size_t)(mts * 16 + lm) * KP;   // M-tile 0 row
    const __bf16* ar1 = ar0 + (size_t)16 * KP;                   // M-tile 1 row
    #pragma unroll 8
    for (int kb = 0; kb < KBLKS; ++kb) {
      v16bf bf = *(const v16bf*)(wbase + (size_t)kb * (64 * 512));
      v8bf a00 = *(const v8bf*)(ar0 + kb * 32 + c0);
      v8bf a01 = *(const v8bf*)(ar0 + kb * 32 + c1);
      v8bf a10 = *(const v8bf*)(ar1 + kb * 32 + c0);
      v8bf a11 = *(const v8bf*)(ar1 + kb * 32 + c1);
      v16bf A0 = SHUF16(a00, a01);
      v16bf A1 = SHUF16(a10, a11);
      acc0 = __builtin_amdgcn_wmma_f32_16x16x32_bf16(false, A0, false, bf, (short)0, acc0, false, false);
      acc1 = __builtin_amdgcn_wmma_f32_16x16x32_bf16(false, A1, false, bf, (short)0, acc1, false, false);
    }

    __bf16* hdst = ((t + 1) & 1) ? hbuf1 : hbuf0;
    #pragma unroll
    for (int r = 0; r < 8; ++r) {
      hdst[(size_t)(mts * 16 + half * 8 + r) * HID + ncol]       = (__bf16)acc0[r];
      hdst[(size_t)((mts + 1) * 16 + half * 8 + r) * HID + ncol] = (__bf16)acc1[r];
    }

    // device-scope grid barrier between sequential steps.
    // release: every thread flushes its own stores to agent scope first.
    __builtin_amdgcn_fence(__ATOMIC_RELEASE, "agent");
    __syncthreads();
    if (tid == 0) {
      atomicAdd(sync, 1u);
      const unsigned target = (unsigned)NWG * (unsigned)(t + 1);
      while (__atomic_load_n(sync, __ATOMIC_ACQUIRE) < target)
        __builtin_amdgcn_s_sleep(1);
    }
    __syncthreads();
    // acquire: invalidate near caches so next staging sees fresh hbuf
    __builtin_amdgcn_fence(__ATOMIC_ACQUIRE, "agent");
  }

  // Final step: out = tanh([x_{T-1}, h_{T-1}] @ W_i2o + b_i2o)
  {
    const int t = T_STEPS - 1;
    const __bf16* hsrc = (t & 1) ? hbuf1 : hbuf0;
    stage_comb(comb, x + (size_t)t * BATCH * IN_DIM, hsrc, tid);
    __syncthreads();

    const int nt2    = wave & 1;                 // 8 waves -> 2 N-tiles x 4 M-tiles
    const int mt     = wave >> 1;
    const int ncol_o = wg * 32 + nt2 * 16 + lm;  // global O column
    const int nblk_o = wg * 2 + nt2;
    const float bvo  = b_o[ncol_o];

    v8f acc;
    #pragma unroll
    for (int r = 0; r < 8; ++r) acc[r] = bvo;

    const __bf16* wobase = wo_sw + ((size_t)nblk_o * 32 + lane) * 16;
    const __bf16* ar0 = comb + (size_t)(mt * 16 + lm) * KP;
    #pragma unroll 8
    for (int kb = 0; kb < KBLKS; ++kb) {
      v16bf bf = *(const v16bf*)(wobase + (size_t)kb * (32 * 512));
      v8bf a00 = *(const v8bf*)(ar0 + kb * 32 + c0);
      v8bf a01 = *(const v8bf*)(ar0 + kb * 32 + c1);
      v16bf A0 = SHUF16(a00, a01);
      acc = __builtin_amdgcn_wmma_f32_16x16x32_bf16(false, A0, false, bf, (short)0, acc, false, false);
    }
    #pragma unroll
    for (int r = 0; r < 8; ++r)
      out[(size_t)(mt * 16 + half * 8 + r) * OUT_DIM + ncol_o] = tanhf(acc[r]);
  }
}

extern "C" void kernel_launch(void* const* d_in, const int* in_sizes, int n_in,
                              void* d_out, int out_size, void* d_ws, size_t ws_size,
                              hipStream_t stream) {
  (void)in_sizes; (void)n_in; (void)out_size; (void)ws_size;
  const float* x  = (const float*)d_in[0];
  const float* Wh = (const float*)d_in[1];
  const float* bh = (const float*)d_in[2];
  const float* Wo = (const float*)d_in[3];
  const float* bo = (const float*)d_in[4];

  char* ws = (char*)d_ws;
  __bf16* wh_sw = (__bf16*)(ws + WS_WH_OFF);
  __bf16* wo_sw = (__bf16*)(ws + WS_WO_OFF);
  __bf16* h0    = (__bf16*)(ws + WS_H0_OFF);
  __bf16* h1    = (__bf16*)(ws + WS_H1_OFF);
  unsigned* syn = (unsigned*)(ws + WS_SYNC_OFF);

  const long prep_n = (long)KTOT * HID + (long)KTOT * OUT_DIM + (long)BATCH * HID + 1;
  rnn_prep<<<(unsigned)((prep_n + 255) / 256), 256, 0, stream>>>(Wh, Wo, wh_sw, wo_sw, h0, syn);

  const size_t lds_bytes = (size_t)BATCH * KP * sizeof(__bf16);   // 197,632 B
  // raising the dynamic-LDS cap is idempotent and not a captured stream op
  hipFuncSetAttribute((const void*)rnn_persist,
                      hipFuncAttributeMaxDynamicSharedMemorySize, (int)lds_bytes);

  rnn_persist<<<NWG, 256, lds_bytes, stream>>>(x, bh, bo, wh_sw, wo_sw, h0, h1, syn,
                                               (float*)d_out);
}
